// MultiQueryAttention_45311904972979
// MI455X (gfx1250) — matrix operand order repaired and checked by reading, not compiled, not run on hardware
//
#include <hip/hip_runtime.h>

// ---------------------------------------------------------------------------
// Problem constants (from reference)
// ---------------------------------------------------------------------------
#define HID 2048
#define NH  16
#define HD  128
#define BB  2
#define SS  2048
#define MM  (BB * SS)     // 4096 flattened tokens

typedef __bf16 bf16_t;
typedef __attribute__((ext_vector_type(16))) __bf16 v16bf;
typedef __attribute__((ext_vector_type(8)))  __bf16 v8bf;
typedef __attribute__((ext_vector_type(8)))  float  v8f;
typedef __attribute__((ext_vector_type(4)))  unsigned int u32x4;
typedef __attribute__((ext_vector_type(8)))  int i32x8;
typedef __attribute__((ext_vector_type(4)))  int i32x4;

#if defined(__gfx1250__) && __has_builtin(__builtin_amdgcn_tensor_load_to_lds)
#define USE_TDM 1
#else
#define USE_TDM 0
#endif

// Build a 16-element bf16 fragment from two 16-byte runs.
static __device__ inline v16bf frag16(const bf16_t* plo, const bf16_t* phi) {
  v8bf a = *(const v8bf*)plo;
  v8bf b = *(const v8bf*)phi;
  v16bf r;
#pragma unroll
  for (int i = 0; i < 8; ++i) { r[i] = a[i]; r[i + 8] = b[i]; }
  return r;
}

static __device__ inline v8f vzero8() {
  v8f r;
#pragma unroll
  for (int i = 0; i < 8; ++i) r[i] = 0.0f;
  return r;
}

#if USE_TDM
// Issue a TDM DMA of a 128-row x 32-col bf16 tile (row stride = stride0
// elements) from global memory into LDS at byte offset lds_off, with LDS
// padding of 4 DWORDs after every 16 DWORDs (row stride 40 bf16 elements).
// Descriptor packing per CDNA5 ISA sec. 8.3 / 8.4 (D# groups 0 and 1).
static __device__ inline void tdm_load_tile_128x32(const bf16_t* gptr,
                                                   unsigned lds_off,
                                                   unsigned tdim0,
                                                   unsigned tdim1,
                                                   unsigned stride0) {
  unsigned long long ga = (unsigned long long)(const void*)gptr;
  u32x4 g0;
  g0[0] = 1u;                                        // count=1, user D#
  g0[1] = lds_off;                                   // LDS byte address
  g0[2] = (unsigned)(ga & 0xffffffffu);              // global addr [31:0]
  g0[3] = (unsigned)((ga >> 32) & 0x01ffffffu)       // global addr [56:32]
          | (2u << 30);                              // type = 2 ("image")
  i32x8 g1;
  g1[0] = (int)((1u << 16)      // data_size = 1 -> 2 bytes (bf16)
                | (1u << 20)    // pad_enable
                | (3u << 22)    // pad_interval code 3 -> every 16 DWORDs
                | (3u << 25));  // pad_amount  code 3 -> 4 DWORDs (16 B)
  g1[1] = (int)((tdim0 & 0xffffu) << 16);                            // tensor_dim0 lo
  g1[2] = (int)(((tdim0 >> 16) & 0xffffu) | ((tdim1 & 0xffffu) << 16));
  g1[3] = (int)(((tdim1 >> 16) & 0xffffu) | (32u << 16));            // tile_dim0 = 32
  g1[4] = 128;                                                       // tile_dim1 = 128
  g1[5] = (int)stride0;                                              // dim0 stride lo
  g1[6] = 0;
  g1[7] = 0;
  i32x4 z4;
  z4[0] = 0; z4[1] = 0; z4[2] = 0; z4[3] = 0;
#if __clang_major__ >= 23
  i32x8 z8;
#pragma unroll
  for (int i = 0; i < 8; ++i) z8[i] = 0;
  __builtin_amdgcn_tensor_load_to_lds(g0, g1, z4, z4, z8, 0);
#else
  __builtin_amdgcn_tensor_load_to_lds(g0, g1, z4, z4, 0);
#endif
}
#endif  // USE_TDM

// ---------------------------------------------------------------------------
// f32 -> bf16 conversion (grid-stride)
// ---------------------------------------------------------------------------
__global__ void f32_to_bf16_kernel(const float* __restrict__ in,
                                   bf16_t* __restrict__ out, long n) {
  long i = (long)blockIdx.x * blockDim.x + threadIdx.x;
  long stride = (long)gridDim.x * blockDim.x;
  for (; i < n; i += stride) out[i] = (bf16_t)in[i];
}

// ---------------------------------------------------------------------------
// Tiled bf16 WMMA GEMM: C[M,N] = A[M,K] @ W[N,K]^T + bias[N]
//   Block tile 128x128, BK=32, 256 threads = 8 waves, wave tile 32x64.
//   Double-buffered LDS; tiles staged by the Tensor Data Mover (wave 0 issues
//   the DMA for buffer kt+1 while all waves run WMMAs on buffer kt), with
//   s_wait_tensorcnt + block barrier for the handoff.
//   Optional outputs: f32 C, bf16 C, bf16 C^T (leading dim M).
// ---------------------------------------------------------------------------
__global__ __launch_bounds__(256) void gemm_bf16_wmma(
    const bf16_t* __restrict__ A,     // [M,K]
    const bf16_t* __restrict__ W,     // [N,K]
    const float*  __restrict__ bias,  // [N]
    float*   __restrict__ outF,       // [M,N] f32   (nullable)
    bf16_t*  __restrict__ outB,       // [M,N] bf16  (nullable)
    bf16_t*  __restrict__ outT,       // [N,M] bf16  (nullable)
    int M, int N, int K) {
  __shared__ alignas(16) bf16_t As[2][128][40];
  __shared__ alignas(16) bf16_t Bs[2][128][40];

  const int mBase = blockIdx.y * 128;
  const int nBase = blockIdx.x * 128;
  const int tid   = threadIdx.x;
  const int wave  = tid >> 5;
  const int lane  = tid & 31;
  const int l16   = lane & 15;
  const int hih   = lane >> 4;           // 0 = lanes 0-15, 1 = lanes 16-31
  const int wm    = (wave & 3) * 32;     // wave M offset in block tile
  const int wn    = (wave >> 2) * 64;    // wave N offset in block tile

  v8f acc[2][4];
#pragma unroll
  for (int i = 0; i < 2; ++i)
#pragma unroll
    for (int j = 0; j < 4; ++j) acc[i][j] = vzero8();

  const int KT = K / 32;

#if USE_TDM
  // Prologue: DMA the first A/W tiles into buffer 0.
  if (wave == 0) {
    tdm_load_tile_128x32(&A[(size_t)mBase * K], (unsigned)(size_t)&As[0][0][0],
                         (unsigned)K, (unsigned)M, (unsigned)K);
    tdm_load_tile_128x32(&W[(size_t)nBase * K], (unsigned)(size_t)&Bs[0][0][0],
                         (unsigned)K, (unsigned)N, (unsigned)K);
    __builtin_amdgcn_s_wait_tensorcnt(0);
  }
  __syncthreads();
#endif

  for (int kt = 0; kt < KT; ++kt) {
    const int cur = kt & 1;
    const int k0  = kt * 32;

#if USE_TDM
    // Kick off the DMA for the next K-step into the other buffer.
    if (wave == 0 && kt + 1 < KT) {
      tdm_load_tile_128x32(&A[(size_t)mBase * K + k0 + 32],
                           (unsigned)(size_t)&As[1 - cur][0][0],
                           (unsigned)K, (unsigned)M, (unsigned)K);
      tdm_load_tile_128x32(&W[(size_t)nBase * K + k0 + 32],
                           (unsigned)(size_t)&Bs[1 - cur][0][0],
                           (unsigned)K, (unsigned)N, (unsigned)K);
    }
#else
    // Cooperative stage of A-tile (128x32) and W-tile (128x32) to LDS.
#pragma unroll
    for (int c = tid; c < 512; c += 256) {
      const int row = c >> 2;
      const int col = (c & 3) * 8;
      *(v8bf*)&As[cur][row][col] =
          *(const v8bf*)&A[(size_t)(mBase + row) * K + k0 + col];
      *(v8bf*)&Bs[cur][row][col] =
          *(const v8bf*)&W[(size_t)(nBase + row) * K + k0 + col];
    }
    __syncthreads();
#endif

    v16bf af[2], wf[4];
#pragma unroll
    for (int i = 0; i < 2; ++i) {
      const int r = wm + i * 16 + l16;
      af[i] = frag16(&As[cur][r][hih ? 16 : 0], &As[cur][r][hih ? 24 : 8]);
    }
#pragma unroll
    for (int j = 0; j < 4; ++j) {
      const int n = wn + j * 16 + l16;
      const bf16_t* p = &Bs[cur][n][hih ? 16 : 0];
      wf[j] = frag16(p, p + 8);
    }
#pragma unroll
    for (int i = 0; i < 2; ++i)
#pragma unroll
      for (int j = 0; j < 4; ++j)
        acc[i][j] = __builtin_amdgcn_wmma_f32_16x16x32_bf16(
            false, af[i], false, wf[j], (short)0, acc[i][j], false, false);

    __syncthreads();  // all waves done reading buffer `cur`

#if USE_TDM
    // Publish the next buffer: wave 0 drains TENSORcnt, then block barrier.
    if (wave == 0 && kt + 1 < KT) __builtin_amdgcn_s_wait_tensorcnt(0);
    __syncthreads();
#endif
  }

  // Epilogue: bias + store(s). C layout: elem v -> row (v or v+8), col l16.
#pragma unroll
  for (int i = 0; i < 2; ++i)
#pragma unroll
    for (int j = 0; j < 4; ++j) {
      const int n = nBase + wn + j * 16 + l16;
      const float bv = bias[n];
#pragma unroll
      for (int v = 0; v < 8; ++v) {
        const int m = mBase + wm + i * 16 + (hih ? v + 8 : v);
        const float val = acc[i][j][v] + bv;
        if (outF) outF[(size_t)m * N + n] = val;
        if (outB) outB[(size_t)m * N + n] = (bf16_t)val;
        if (outT) outT[(size_t)n * M + m] = (bf16_t)val;
      }
    }
}

// ---------------------------------------------------------------------------
// Causal MQA attention (single KV head), flash-style online softmax.
//   One wave owns 16 query rows of one (b, h). Streams 32 keys / iteration:
//   8 WMMA for S = Q K^T (Kdim=128), LDS transpose of P, 8 WMMA for O += P V.
//   Output written directly in the reference's transposed layout:
//   Oout[b][h*128 + d][s] = aw[b, h, s, d]   (H*D == S == 2048)
// ---------------------------------------------------------------------------
__global__ __launch_bounds__(256) void mqa_attn_wmma(
    const bf16_t* __restrict__ Q,    // [B*S, 2048]
    const bf16_t* __restrict__ Kb,   // [B*S, 128]
    const bf16_t* __restrict__ Vt,   // [128, B*S]  (d-major)
    bf16_t* __restrict__ Oout) {     // [B, 2048, 2048]
  __shared__ alignas(16) bf16_t Pb[8][16][40];  // per-wave P transpose buffer

  const int wave = threadIdx.x >> 5;
  const int lane = threadIdx.x & 31;
  const int l16  = lane & 15;
  const int hih  = lane >> 4;

  const int task = blockIdx.x * 8 + wave;       // [0, B*NH*(S/16))
  const int qt   = task & 127;                  // S/16 = 128 query tiles
  const int h    = (task >> 7) & (NH - 1);
  const int b    = task >> 11;
  if (b >= BB) return;
  const int qBase = qt * 16;

  // Q fragments for this wave's 16 rows (Kdim = 128 -> 4 fragments).
  v16bf qf[4];
  const bf16_t* qrow = Q + (size_t)(b * SS + qBase + l16) * HID + h * HD;
#pragma unroll
  for (int dt = 0; dt < 4; ++dt)
    qf[dt] = frag16(qrow + dt * 32 + (hih ? 16 : 0),
                    qrow + dt * 32 + (hih ? 24 : 8));

  v8f o[8];
#pragma unroll
  for (int n = 0; n < 8; ++n) o[n] = vzero8();
  v8f mrow, lrow;
#pragma unroll
  for (int v = 0; v < 8; ++v) { mrow[v] = -1e30f; lrow[v] = 0.0f; }

  const float scale = 0.08838834764831845f;     // 1/sqrt(128)
  const int kIters = (qBase + 16 + 31) / 32;    // causal bound, 32 keys/iter

  for (int it = 0; it < kIters; ++it) {
    const int kBase = it * 32;

    // ---- scores: two 16-key tiles, reduce over D=128 ----
    v8f sc[2];
    sc[0] = vzero8(); sc[1] = vzero8();
#pragma unroll
    for (int t = 0; t < 2; ++t)
#pragma unroll
      for (int dt = 0; dt < 4; ++dt) {
        const bf16_t* kp = Kb + (size_t)(b * SS + kBase + t * 16 + l16) * HD +
                           dt * 32 + (hih ? 16 : 0);
        v16bf kf = frag16(kp, kp + 8);
        sc[t] = __builtin_amdgcn_wmma_f32_16x16x32_bf16(
            false, qf[dt], false, kf, (short)0, sc[t], false, false);
      }

    // ---- scale + causal mask ----
#pragma unroll
    for (int t = 0; t < 2; ++t)
#pragma unroll
      for (int v = 0; v < 8; ++v) {
        const int r = qBase + (hih ? v + 8 : v);
        const int c = kBase + t * 16 + l16;
        float s = sc[t][v] * scale;
        if (c > r) s = -1e30f;
        sc[t][v] = s;
      }

    // ---- online softmax: row max / exp / row sum (16-lane groups) ----
    v8f mnew, rsum;
#pragma unroll
    for (int v = 0; v < 8; ++v) {
      float mx = fmaxf(sc[0][v], sc[1][v]);
#pragma unroll
      for (int off = 1; off < 16; off <<= 1)
        mx = fmaxf(mx, __shfl_xor(mx, off, 16));
      mnew[v] = fmaxf(mrow[v], mx);
    }
#pragma unroll
    for (int v = 0; v < 8; ++v) {
      float p0 = __expf(sc[0][v] - mnew[v]);
      float p1 = __expf(sc[1][v] - mnew[v]);
      sc[0][v] = p0; sc[1][v] = p1;
      float rs = p0 + p1;
#pragma unroll
      for (int off = 1; off < 16; off <<= 1)
        rs += __shfl_xor(rs, off, 16);
      rsum[v] = rs;
    }
#pragma unroll
    for (int v = 0; v < 8; ++v) {
      const float corr = __expf(mrow[v] - mnew[v]);
      lrow[v] = lrow[v] * corr + rsum[v];
#pragma unroll
      for (int n = 0; n < 8; ++n) o[n][v] *= corr;
      mrow[v] = mnew[v];
    }

    // ---- transpose P (C-layout -> A-fragment layout) through LDS ----
#pragma unroll
    for (int t = 0; t < 2; ++t)
#pragma unroll
      for (int v = 0; v < 8; ++v) {
        const int r = hih ? v + 8 : v;
        Pb[wave][r][t * 16 + l16] = (bf16_t)sc[t][v];
      }
    __threadfence_block();  // order ds stores before ds loads (wave-private)
    const bf16_t* prow = &Pb[wave][l16][0];
    v16bf pf = frag16(prow + (hih ? 16 : 0), prow + (hih ? 24 : 8));

    // ---- O += P @ V  (B-frags contiguous thanks to transposed Vt) ----
#pragma unroll
    for (int n = 0; n < 8; ++n) {
      const bf16_t* vp = Vt + (size_t)(n * 16 + l16) * MM + b * SS + kBase +
                         (hih ? 16 : 0);
      v16bf vf = frag16(vp, vp + 8);
      o[n] = __builtin_amdgcn_wmma_f32_16x16x32_bf16(
          false, pf, false, vf, (short)0, o[n], false, false);
    }
  }

  // ---- epilogue: normalize and store transposed (i = h*128+d, col = s) ----
#pragma unroll
  for (int n = 0; n < 8; ++n)
#pragma unroll
    for (int v = 0; v < 8; ++v) {
      const int d = n * 16 + l16;
      const int m = qBase + (hih ? v + 8 : v);
      const float val = o[n][v] / lrow[v];
      Oout[(size_t)b * HID * SS + (size_t)(h * HD + d) * SS + m] = (bf16_t)val;
    }
}

// ---------------------------------------------------------------------------
// Host-side orchestration
// ---------------------------------------------------------------------------
extern "C" void kernel_launch(void* const* d_in, const int* in_sizes, int n_in,
                              void* d_out, int out_size, void* d_ws,
                              size_t ws_size, hipStream_t stream) {
  const float* x    = (const float*)d_in[0];
  // d_in[1] = attention_mask (causal; applied analytically)
  const float* qw_w = (const float*)d_in[2];
  const float* qw_b = (const float*)d_in[3];
  const float* kw_w = (const float*)d_in[4];
  const float* kw_b = (const float*)d_in[5];
  const float* vw_w = (const float*)d_in[6];
  const float* vw_b = (const float*)d_in[7];
  const float* ow_w = (const float*)d_in[8];
  const float* ow_b = (const float*)d_in[9];

  char* ws = (char*)d_ws;
  size_t off = 0;
  auto carve = [&](size_t bytes) {
    void* p = ws + off;
    off += (bytes + 255) & ~(size_t)255;
    return p;
  };
  bf16_t* xb  = (bf16_t*)carve((size_t)MM * HID * 2);   // x, bf16
  bf16_t* qwb = (bf16_t*)carve((size_t)HID * HID * 2);  // qw_w, bf16
  bf16_t* kwb = (bf16_t*)carve((size_t)HD * HID * 2);   // kw_w, bf16
  bf16_t* vwb = (bf16_t*)carve((size_t)HD * HID * 2);   // vw_w, bf16
  bf16_t* owb = (bf16_t*)carve((size_t)HID * HID * 2);  // ow_w, bf16
  bf16_t* qb  = (bf16_t*)carve((size_t)MM * HID * 2);   // Q proj, bf16
  bf16_t* kb  = (bf16_t*)carve((size_t)MM * HD * 2);    // K proj, bf16
  bf16_t* vtb = (bf16_t*)carve((size_t)HD * MM * 2);    // V proj transposed
  bf16_t* ao  = (bf16_t*)carve((size_t)MM * HID * 2);   // attn out (transposed layout)
  (void)ws_size; (void)in_sizes; (void)n_in; (void)out_size;

  auto cvt = [&](const float* in, bf16_t* out, long n) {
    int blocks = (int)((n + 1023) / 1024);
    if (blocks > 16384) blocks = 16384;
    f32_to_bf16_kernel<<<blocks, 256, 0, stream>>>(in, out, n);
  };
  cvt(x,    xb,  (long)MM * HID);
  cvt(qw_w, qwb, (long)HID * HID);
  cvt(kw_w, kwb, (long)HD * HID);
  cvt(vw_w, vwb, (long)HD * HID);
  cvt(ow_w, owb, (long)HID * HID);

  // Q projection: [4096,2048] = xb @ qwb^T + qw_b   -> bf16
  gemm_bf16_wmma<<<dim3(HID / 128, MM / 128), 256, 0, stream>>>(
      xb, qwb, qw_b, nullptr, qb, nullptr, MM, HID, HID);
  // K projection: [4096,128] -> bf16
  gemm_bf16_wmma<<<dim3(HD / 128, MM / 128), 256, 0, stream>>>(
      xb, kwb, kw_b, nullptr, kb, nullptr, MM, HD, HID);
  // V projection: [4096,128] -> bf16 transposed [128,4096] for PV B-frags
  gemm_bf16_wmma<<<dim3(HD / 128, MM / 128), 256, 0, stream>>>(
      xb, vwb, vw_b, nullptr, nullptr, vtb, MM, HD, HID);

  // Attention: B*NH*(S/16) = 4096 wave-tasks, 8 waves/block -> 512 blocks.
  mqa_attn_wmma<<<dim3(BB * NH * (SS / 16) / 8), 256, 0, stream>>>(
      qb, kb, vtb, ao);

  // Output projection: d_out[4096,2048] f32 = ao @ owb^T + ow_b
  gemm_bf16_wmma<<<dim3(HID / 128, MM / 128), 256, 0, stream>>>(
      ao, owb, ow_b, (float*)d_out, nullptr, nullptr, MM, HID, HID);
}
